// pFF_46583215293129
// MI455X (gfx1250) — matrix operand order repaired and checked by reading, not compiled, not run on hardware
//
#include <hip/hip_runtime.h>
#include <hip/hip_bf16.h>

typedef __attribute__((ext_vector_type(16))) _Float16 v16h;
typedef __attribute__((ext_vector_type(8)))  float    v8f;

#define H_ 256
#define W_ 256
#define HW_ 65536
#define NB_ 4
#define NC_ 64

__device__ __forceinline__ int refl(int i, int n) {
    // reflect padding, pad=1: -1 -> 1, n -> n-2
    return i < 0 ? -i : (i >= n ? 2 * n - 2 - i : i);
}

// ---------------------------------------------------------------------------
// Kernel 0: pack conv weights w[9,64,3,3] (f32) into CDNA5 WMMA B-fragment
// layout (f16). 18 K-chunks of 32; chunk k covers tap = k>>1, channels
// c0 = (k&1)*32 .. +31.
// B operand layout (16-bit, 32x16, 8 VGPRs, contiguous K order, same as the A operand):
//   lane = 16*half + n (n = column), element j of the v16h maps to
//   kk = j + 16*half   (lanes 0-15: K=0..15 in order, lanes 16-31: K=16..31)
//   value = (n<9) ? w[n, c0+kk, tap/3, tap%3] : 0
// stored at wsB[k*512 + lane*16 + j]  (one contiguous 32B v16h per lane).
// ---------------------------------------------------------------------------
__global__ __launch_bounds__(256) void pFF_pack_w(const float* __restrict__ w,
                                                  _Float16* __restrict__ wsB) {
    int idx = blockIdx.x * 256 + threadIdx.x;          // 0 .. 18*512-1
    if (idx >= 18 * 512) return;
    int k    = idx >> 9;
    int lane = (idx >> 4) & 31;
    int j    = idx & 15;
    int half = lane >> 4;
    int n    = lane & 15;
    int tap  = k >> 1;
    int c0   = (k & 1) * 32;
    int kk   = j + half * 16;             // B-operand K mapping
    int c    = c0 + kk;
    int ki   = tap / 3, kj = tap % 3;
    float v = 0.0f;
    if (n < 9) v = w[((n * NC_ + c) * 3 + ki) * 3 + kj];
    wsB[idx] = (_Float16)v;
}

// ---------------------------------------------------------------------------
// Kernel 1: fused  conv(3x3,reflect) -> +bias -> softmax(9) -> per-pixel
// filter applied to inpt over 64 channels.  One wave32 per 16-pixel row tile.
// ---------------------------------------------------------------------------
__global__ __launch_bounds__(256) void pFF_fused(const float* __restrict__ features,
                                                 const float* __restrict__ inpt,
                                                 const _Float16* __restrict__ wsB,
                                                 const float* __restrict__ bias,
                                                 float* __restrict__ out) {
    __shared__ float logits[8][16][16];   // [wave][pixel][col]
    __shared__ float ffw[8][16][9];       // [wave][pixel][tap]

    const int tid  = threadIdx.x;
    const int wv   = tid >> 5;
    const int lane = tid & 31;
    const int half = lane >> 4;
    const int p    = lane & 15;           // pixel index (A-row / store lane)

    const int t  = blockIdx.x * 8 + wv;   // tile id, 0..16383
    const int b  = t >> 12;               // 4096 tiles per batch image
    const int r2 = t & 4095;
    const int h  = r2 >> 4;
    const int w0 = (r2 & 15) * 16;

    const float* fB = features + (size_t)b * NC_ * HW_;

    // ---- implicit-GEMM conv: M=16 px, N=16 (9 live), K=576, 18 WMMAs ----
    // A operand (16-bit, 16x32): lane = 16*half + p, element j maps to
    //   kk = j + 16*half   (contiguous, matching packed B)
    v8f acc = {};
    #pragma unroll
    for (int tap = 0; tap < 9; ++tap) {
        const int di = tap / 3 - 1, dj = tap % 3 - 1;
        const int rh = refl(h + di, H_);
        const int rw = refl(w0 + p + dj, W_);
        const float* fbase = fB + rh * W_ + rw;
        #pragma unroll
        for (int cc = 0; cc < 2; ++cc) {
            const int c0 = cc * 32;
            v16h a;
            #pragma unroll
            for (int j = 0; j < 16; ++j) {
                int kk = j + half * 16;   // A-operand K mapping
                a[j] = (_Float16)fbase[(size_t)(c0 + kk) * HW_];
            }
            v16h bf = *(const v16h*)(wsB + (size_t)((tap * 2 + cc) * 512 + lane * 16));
            acc = __builtin_amdgcn_wmma_f32_16x16x32_f16(
                false, a, false, bf, (short)0, acc, false, false);
        }
    }

    // ---- spill C (lane = col o; VGPR r = pixel r + 8*half), add bias ----
    const int o   = lane & 15;
    const float bo = (o < 9) ? bias[o] : 0.0f;
    #pragma unroll
    for (int r = 0; r < 8; ++r)
        logits[wv][r + 8 * half][o] = acc[r] + bo;
    __syncthreads();

    // ---- softmax over the 9 taps (lanes 0..15, one pixel each) ----
    if (lane < 16) {
        float v[9];
        float m = -3.0e38f;
        #pragma unroll
        for (int k = 0; k < 9; ++k) { v[k] = logits[wv][lane][k]; m = fmaxf(m, v[k]); }
        float s = 0.0f;
        #pragma unroll
        for (int k = 0; k < 9; ++k) { v[k] = __expf(v[k] - m); s += v[k]; }
        const float inv = 1.0f / s;
        #pragma unroll
        for (int k = 0; k < 9; ++k) ffw[wv][lane][k] = v[k] * inv;
    }
    __syncthreads();

    // ---- apply per-pixel filter to inpt: 32 channels per lane-half (half 0 even, half 1 odd) ----
    float f[9];
    #pragma unroll
    for (int k = 0; k < 9; ++k) f[k] = ffw[wv][p][k];

    const int pw  = w0 + p;
    const int rh0 = refl(h - 1, H_), rh2 = refl(h + 1, H_);
    const int rw0 = refl(pw - 1, W_), rw2 = refl(pw + 1, W_);
    const float* iB = inpt + (size_t)b * NC_ * HW_;
    float* oB = out + (size_t)b * NC_ * HW_ + h * W_ + pw;

    for (int c = half; c < NC_; c += 2) {
        const float* ib = iB + (size_t)c * HW_;
        const float* r0  = ib + rh0 * W_;
        const float* r1  = ib + h   * W_;
        const float* r2p = ib + rh2 * W_;
        float s = r0[rw0]  * f[0] + r0[pw]  * f[1] + r0[rw2]  * f[2]
                + r1[rw0]  * f[3] + r1[pw]  * f[4] + r1[rw2]  * f[5]
                + r2p[rw0] * f[6] + r2p[pw] * f[7] + r2p[rw2] * f[8];
        oB[(size_t)c * HW_] = s;
    }
}

extern "C" void kernel_launch(void* const* d_in, const int* in_sizes, int n_in,
                              void* d_out, int out_size, void* d_ws, size_t ws_size,
                              hipStream_t stream) {
    const float* features = (const float*)d_in[0];  // [4,64,256,256]
    const float* inpt     = (const float*)d_in[1];  // [4,64,256,256]
    const float* w        = (const float*)d_in[2];  // [9,64,3,3]
    const float* bias     = (const float*)d_in[3];  // [9]
    float*       out      = (float*)d_out;          // [4,64,256,256]
    _Float16*    wsB      = (_Float16*)d_ws;        // 18*512 halves = 18 KB

    // pack weights into WMMA B-fragment order (18 KB, L2-resident)
    pFF_pack_w<<<(18 * 512 + 255) / 256, 256, 0, stream>>>(w, wsB);

    // 16384 16-pixel tiles, 8 waves (tiles) per 256-thread block
    pFF_fused<<<2048, 256, 0, stream>>>(features, inpt, wsB, bias, out);
}
